// BahdanauAttention_45165876085309
// MI455X (gfx1250) — matrix operand order repaired and checked by reading, not compiled, not run on hardware
//
#include <hip/hip_runtime.h>
#include <hip/hip_bf16.h>
#include <math.h>

// ---------------------------------------------------------------------------
// Bahdanau attention, fused for MI455X (gfx1250, wave32, WMMA bf16)
//   B=32, L=2048, E=D=A=1024
// Dominant GEMM (B*L x E) @ (E x A) fused with tanh + v-dot epilogue so the
// 256MB [B,L,A] intermediate is never materialized. GEMM runs on
// v_wmma_f32_16x16x32_bf16, W1 pre-converted to bf16 (2MB, L2-resident),
// encoder tiles converted to bf16 in LDS per workgroup.
// Each wave processes 4 N-tiles sharing one A-fragment: 2.5 b128 loads per
// WMMA and 4 independent accumulation chains.
// ---------------------------------------------------------------------------

#define B_DIM 32
#define L_DIM 2048
#define E_DIM 1024
#define A_DIM 1024

typedef __attribute__((ext_vector_type(16))) __bf16 v16bf;
typedef __attribute__((ext_vector_type(8)))  __bf16 v8bf;
typedef __attribute__((ext_vector_type(8)))  float  v8f;
typedef __attribute__((ext_vector_type(4)))  float  v4f;

// Branch-free tanh: 1 - 2/(1+e^{2x}); saturates correctly via exp inf/0.
__device__ __forceinline__ float fast_tanh(float x) {
    float e = __expf(2.0f * x);
    return 1.0f - 2.0f * __builtin_amdgcn_rcpf(1.0f + e);
}

// ---------------- P0a: W1 fp32 -> bf16 (kept in ws, L2-resident: 2MB) ------
__global__ __launch_bounds__(256) void w1_to_bf16(const float* __restrict__ W1,
                                                  __bf16* __restrict__ W1b, int n) {
    int i = blockIdx.x * 256 + threadIdx.x;
    int stride = gridDim.x * 256;
    for (; i < n; i += stride) W1b[i] = (__bf16)W1[i];
}

// ---------------- P0b: decT[b][a] = dot(W2[a,:], dh[b,:]) ------------------
__global__ __launch_bounds__(256) void dec_proj(const float* __restrict__ W2,
                                                const float* __restrict__ dh,
                                                float* __restrict__ decT) {
    __shared__ float sDh[E_DIM];
    int b = blockIdx.x;
    for (int i = threadIdx.x; i < E_DIM; i += 256) sDh[i] = dh[b * E_DIM + i];
    __syncthreads();
    for (int a = threadIdx.x; a < A_DIM; a += 256) {
        const v4f* w = (const v4f*)(W2 + (size_t)a * E_DIM);
        float acc = 0.f;
#pragma unroll 4
        for (int d4 = 0; d4 < E_DIM / 4; ++d4) {
            v4f wv = w[d4];
            acc += wv[0] * sDh[4 * d4 + 0] + wv[1] * sDh[4 * d4 + 1] +
                   wv[2] * sDh[4 * d4 + 2] + wv[3] * sDh[4 * d4 + 3];
        }
        decT[b * A_DIM + a] = acc;
    }
}

// ---------------- P1: fused scores = v . tanh(W1@enc + decT), masked -------
// One workgroup = one batch b, 16 rows of L.  8 waves; each wave covers
// 8 N-tiles as 2 passes of 4 tiles with a shared A-fragment.
__global__ __launch_bounds__(256) void attn_scores(const float* __restrict__ enc,   // [B,L,E]
                                                   const __bf16* __restrict__ W1b,  // [A,E] bf16
                                                   const float* __restrict__ decT,  // [B,A]
                                                   const float* __restrict__ v,     // [A]
                                                   const int* __restrict__ mask,    // [B,L]
                                                   float* __restrict__ scores) {    // [B,L]
    __shared__ __bf16 sA[16][E_DIM];   // 32 KB encoder tile, bf16
    __shared__ float  sScore[16];

    int wg  = blockIdx.x;
    int b   = wg >> 7;                 // 128 L-tiles per batch
    int l0  = (wg & 127) << 4;
    int tid = threadIdx.x;

    // Stage 16 x 1024 encoder rows, converting fp32 -> bf16
    const float* encBase = enc + ((size_t)b * L_DIM + l0) * E_DIM;
    for (int i = tid; i < 16 * E_DIM; i += 256)
        sA[i >> 10][i & (E_DIM - 1)] = (__bf16)encBase[i];
    if (tid < 16) sScore[tid] = 0.f;
    __syncthreads();

    int wave = tid >> 5;
    int lane = tid & 31;
    int m    = lane & 15;   // A-frag row / B-frag column / C-frag N index
    int h    = lane >> 4;   // K-half selector

#pragma unroll
    for (int pass = 0; pass < 2; ++pass) {
        int nbase = wave * 128 + pass * 64;   // first N of the 4-tile group
        // B columns for tiles j=0..3 at wbase + j*16*E_DIM (32KB apart)
        const __bf16* wbase = W1b + (size_t)(nbase + m) * E_DIM;

        v8f cacc[4];
#pragma unroll
        for (int j = 0; j < 4; ++j) cacc[j] = (v8f){};

#pragma unroll 2
        for (int k0 = 0; k0 < E_DIM; k0 += 32) {
            // Shared A fragment (16x32 bf16) from LDS
            const __bf16* ap = &sA[m][k0 + h * 16];
            v8bf a0 = *(const v8bf*)ap;          // K = k0+16h   .. +7
            v8bf a1 = *(const v8bf*)(ap + 8);    // K = k0+16h+8 .. +7
            v16bf afrag;
#pragma unroll
            for (int i = 0; i < 8; ++i) { afrag[i] = a0[i]; afrag[i + 8] = a1[i]; }

            // 4 B fragments (32x16 bf16), 32 contiguous bytes each
            v16bf bf0 = *(const v16bf*)(wbase +  0 * E_DIM + k0 + h * 16);
            v16bf bf1 = *(const v16bf*)(wbase + 16 * E_DIM + k0 + h * 16);
            v16bf bf2 = *(const v16bf*)(wbase + 32 * E_DIM + k0 + h * 16);
            v16bf bf3 = *(const v16bf*)(wbase + 48 * E_DIM + k0 + h * 16);

            cacc[0] = __builtin_amdgcn_wmma_f32_16x16x32_bf16(
                          false, afrag, false, bf0, (short)0, cacc[0], false, false);
            cacc[1] = __builtin_amdgcn_wmma_f32_16x16x32_bf16(
                          false, afrag, false, bf1, (short)0, cacc[1], false, false);
            cacc[2] = __builtin_amdgcn_wmma_f32_16x16x32_bf16(
                          false, afrag, false, bf2, (short)0, cacc[2], false, false);
            cacc[3] = __builtin_amdgcn_wmma_f32_16x16x32_bf16(
                          false, afrag, false, bf3, (short)0, cacc[3], false, false);
        }

        // Epilogue per tile: C[M,N], N = m for this lane, M = r + 8*h
#pragma unroll
        for (int j = 0; j < 4; ++j) {
            int n0 = nbase + j * 16;
            float dv = decT[b * A_DIM + n0 + m];
            float vw = v[n0 + m];
#pragma unroll
            for (int r = 0; r < 8; ++r) {
                float t = vw * fast_tanh(cacc[j][r] + dv);
                // reduce over the 16 N-lanes within this half
                t += __shfl_xor(t, 1, 32);
                t += __shfl_xor(t, 2, 32);
                t += __shfl_xor(t, 4, 32);
                t += __shfl_xor(t, 8, 32);
                if (m == 0) atomicAdd(&sScore[r + 8 * h], t);   // ds_add_f32
            }
        }
    }
    __syncthreads();

    if (tid < 16) {
        int l = l0 + tid;
        float s = sScore[tid];
        scores[b * L_DIM + l] = mask[b * L_DIM + l] ? s : -INFINITY;
    }
}

// ---------------- P2: row softmax -> attn ----------------------------------
__global__ __launch_bounds__(256) void softmax_rows(const float* __restrict__ scores,
                                                    float* __restrict__ attn) {
    __shared__ float red[256];
    int b = blockIdx.x, tid = threadIdx.x;
    const float* row = scores + (size_t)b * L_DIM;
    float vals[8];
    float lm = -INFINITY;
#pragma unroll
    for (int j = 0; j < 8; ++j) { vals[j] = row[tid + j * 256]; lm = fmaxf(lm, vals[j]); }
    red[tid] = lm; __syncthreads();
    for (int s = 128; s > 0; s >>= 1) {
        if (tid < s) red[tid] = fmaxf(red[tid], red[tid + s]);
        __syncthreads();
    }
    float mx = red[0]; __syncthreads();
    float ls = 0.f;
#pragma unroll
    for (int j = 0; j < 8; ++j) { vals[j] = __expf(vals[j] - mx); ls += vals[j]; }
    red[tid] = ls; __syncthreads();
    for (int s = 128; s > 0; s >>= 1) {
        if (tid < s) red[tid] += red[tid + s];
        __syncthreads();
    }
    float inv = 1.f / red[0];
#pragma unroll
    for (int j = 0; j < 8; ++j) attn[(size_t)b * L_DIM + tid + j * 256] = vals[j] * inv;
}

// ---------------- P3: context = attn^T @ enc -------------------------------
__global__ __launch_bounds__(256) void ctx_zero(float* __restrict__ ctx) {
    int i = blockIdx.x * 256 + threadIdx.x;
    if (i < B_DIM * E_DIM) ctx[i] = 0.f;
}

__global__ __launch_bounds__(256) void ctx_accum(const float* __restrict__ enc,
                                                 const float* __restrict__ attn,
                                                 float* __restrict__ ctx) {
    __shared__ float sAttn[512];
    int blk = blockIdx.x;          // b*16 + ec*4 + lc
    int b  = blk >> 4;
    int ec = (blk >> 2) & 3;
    int lc = blk & 3;
    int l0 = lc * 512;
    for (int i = threadIdx.x; i < 512; i += 256)
        sAttn[i] = attn[(size_t)b * L_DIM + l0 + i];
    __syncthreads();
    int e = ec * 256 + threadIdx.x;
    const float* ep = enc + ((size_t)b * L_DIM + l0) * E_DIM + e;
    float acc = 0.f;
#pragma unroll 4
    for (int l = 0; l < 512; ++l) acc += sAttn[l] * ep[(size_t)l * E_DIM];
    atomicAdd(&ctx[b * E_DIM + e], acc);   // global_atomic_add_f32
}

// ---------------------------------------------------------------------------
extern "C" void kernel_launch(void* const* d_in, const int* in_sizes, int n_in,
                              void* d_out, int out_size, void* d_ws, size_t ws_size,
                              hipStream_t stream) {
    (void)in_sizes; (void)n_in; (void)out_size; (void)ws_size;
    const float* enc  = (const float*)d_in[0];   // [B,L,E]
    const float* dh   = (const float*)d_in[1];   // [B,D]
    const int*   mask = (const int*)d_in[2];     // [B,L]
    const float* W1   = (const float*)d_in[3];   // [A,E]
    const float* W2   = (const float*)d_in[4];   // [A,D]
    const float* vv   = (const float*)d_in[5];   // [1,A]

    float* out  = (float*)d_out;
    float* ctx  = out;                    // [B,E], first in tuple
    float* attn = out + B_DIM * E_DIM;    // [B,L], second in tuple

    char* ws = (char*)d_ws;
    __bf16* W1b   = (__bf16*)ws;                                   // 2 MB
    float*  decT  = (float*)(ws + (size_t)A_DIM * E_DIM * 2);      // 128 KB
    float*  score = (float*)(ws + (size_t)A_DIM * E_DIM * 2
                                + (size_t)B_DIM * A_DIM * 4);      // 256 KB

    w1_to_bf16 <<<1024, 256, 0, stream>>>(W1, W1b, A_DIM * E_DIM);
    dec_proj   <<<B_DIM, 256, 0, stream>>>(W2, dh, decT);
    attn_scores<<<B_DIM * (L_DIM / 16), 256, 0, stream>>>(enc, W1b, decT, vv, mask, score);
    softmax_rows<<<B_DIM, 256, 0, stream>>>(score, attn);
    ctx_zero   <<<(B_DIM * E_DIM + 255) / 256, 256, 0, stream>>>(ctx);
    ctx_accum  <<<B_DIM * 16, 256, 0, stream>>>(enc, attn, ctx);
}
